// EncCelebAGCN_52793738002763
// MI455X (gfx1250) — matrix operand order repaired and checked by reading, not compiled, not run on hardware
//
#include <hip/hip_runtime.h>

typedef __attribute__((ext_vector_type(16))) _Float16 v16h;
typedef __attribute__((ext_vector_type(8)))  float    v8f;

// ---------------- utility kernels ----------------

__global__ void gcn_zero_kernel(float* __restrict__ p, long long n) {
  long long i = (long long)blockIdx.x * blockDim.x + threadIdx.x;
  long long stride = (long long)gridDim.x * blockDim.x;
  for (; i < n; i += stride) p[i] = 0.0f;
}

__global__ void gcn_degree_kernel(const int* __restrict__ src, const int* __restrict__ dst,
                                  float* __restrict__ deg_out, float* __restrict__ deg_in,
                                  int E) {
  int e = blockIdx.x * blockDim.x + threadIdx.x;
  if (e >= E) return;
  unsafeAtomicAdd(&deg_out[src[e]], 1.0f);
  unsafeAtomicAdd(&deg_in[dst[e]], 1.0f);
}

__global__ void gcn_norm_kernel(float* __restrict__ p, int n) {
  int i = blockIdx.x * blockDim.x + threadIdx.x;
  if (i >= n) return;
  p[i] = rsqrtf(fmaxf(p[i], 1.0f));
}

// h[n, :] = x[n, :] * norm[n]   (d = 1<<dshift elements per row)
__global__ void gcn_scale_kernel(const float* __restrict__ x, const float* __restrict__ norm,
                                 float* __restrict__ h, int n_elems, int dshift) {
  int i = blockIdx.x * blockDim.x + threadIdx.x;
  if (i >= n_elems) return;
  h[i] = x[i] * norm[i >> dshift];
}

// agg[dst[e], :] += h[src[e], :], one thread per (edge, 4-float chunk)
__global__ void gcn_scatter_kernel(const float* __restrict__ h, const int* __restrict__ src,
                                   const int* __restrict__ dst, float* __restrict__ agg,
                                   long long total, int cshift, int d) {
  long long t = (long long)blockIdx.x * blockDim.x + threadIdx.x;
  if (t >= total) return;
  int e = (int)(t >> cshift);
  int c = ((int)t) & ((1 << cshift) - 1);
  int s  = src[e];
  int dd = dst[e];
  const float4 v = *(const float4*)(h + (size_t)s * d + c * 4);
  float* a = agg + (size_t)dd * d + c * 4;
  unsafeAtomicAdd(a + 0, v.x);
  unsafeAtomicAdd(a + 1, v.y);
  unsafeAtomicAdd(a + 2, v.z);
  unsafeAtomicAdd(a + 3, v.w);
}

// ---------------- WMMA GEMM:  out = (A * norm_dst[row]) @ W + bias ----------------
// A: [N, K] f32 (row major), W: [K, NOUT] f32 (row major), out: [N, NOUT] f32.
// One wave per 16-row M-tile; loops over NOUT/16 column tiles, K/32 WMMA steps.
template <int K, int NOUT>
__global__ __launch_bounds__(256)
void gcn_gemm_wmma_kernel(const float* __restrict__ A, const float* __restrict__ norm_dst,
                          const float* __restrict__ W, const float* __restrict__ bias,
                          float* __restrict__ out, int nMTiles) {
  int wave = (int)((blockIdx.x * blockDim.x + threadIdx.x) >> 5);
  if (wave >= nMTiles) return;                 // whole-wave exit: EXEC stays all-1s for WMMA
  int lane = threadIdx.x & 31;
  int m    = lane & 15;                        // A-matrix row within tile / B,C,D column
  int half = lane >> 4;                        // K-half selector (A) / row+8 selector (C,D)

  int row = wave * 16 + m;
  float ns = norm_dst[row];
  const float* arow = A + (size_t)row * K;

  constexpr int NKT = K / 32;
  v16h afrag[NKT];
  // A fragment layout (16-bit A 16x32): VGPR v<4 -> K = 2v + 8*half (+1),
  //                                     VGPR v>=4 -> K = 16 + 2(v-4) + 8*half (+1)
#pragma unroll
  for (int kt = 0; kt < NKT; ++kt) {
    const float* ak = arow + kt * 32;
#pragma unroll
    for (int v = 0; v < 8; ++v) {
      int k0 = ((v < 4) ? 2 * v : 16 + 2 * (v - 4)) + 8 * half;
      afrag[kt][2 * v]     = (_Float16)(ak[k0] * ns);
      afrag[kt][2 * v + 1] = (_Float16)(ak[k0 + 1] * ns);
    }
  }

#pragma unroll
  for (int nt = 0; nt < NOUT / 16; ++nt) {
    v8f c = {};
#pragma unroll
    for (int kt = 0; kt < NKT; ++kt) {
      // B fragment: lane column n = nt*16 + m; VGPR v<4 -> K = kt*32 + 8*half + 2v (+1), v>=4 -> K = kt*32 + 16 + 8*half + 2(v-4) (+1)
      v16h b;
      const float* wk = W + (size_t)(kt * 32 + 8 * half) * NOUT + nt * 16 + m;
#pragma unroll
      for (int v = 0; v < 8; ++v) {
        b[2 * v]     = (_Float16)wk[((v < 4) ? 2 * v : 2 * v + 8) * NOUT];
        b[2 * v + 1] = (_Float16)wk[((v < 4) ? 2 * v + 1 : 2 * v + 9) * NOUT];
      }
      c = __builtin_amdgcn_wmma_f32_16x16x32_f16(
          /*neg_a=*/false, afrag[kt], /*neg_b=*/false, b,
          /*c_mod=*/(short)0, c, /*reuse_a=*/false, /*reuse_b=*/false);
    }
    // C/D layout: VGPR r -> row r + 8*half, col = m
    float bv = bias[nt * 16 + m];
    float* obase = out + (size_t)(wave * 16) * NOUT + nt * 16 + m;
#pragma unroll
    for (int r = 0; r < 8; ++r) {
      obase[(size_t)(r + 8 * half) * NOUT] = c[r] + bv;
    }
  }
}

// ---------------- launcher ----------------

extern "C" void kernel_launch(void* const* d_in, const int* in_sizes, int n_in,
                              void* d_out, int out_size, void* d_ws, size_t ws_size,
                              hipStream_t stream) {
  const float* x  = (const float*)d_in[0];
  const int*   src = (const int*)d_in[1];
  const int*   dst = (const int*)d_in[2];
  const float* W1 = (const float*)d_in[3];
  const float* b1 = (const float*)d_in[4];
  const float* W2 = (const float*)d_in[5];
  const float* b2 = (const float*)d_in[6];
  float* out = (float*)d_out;

  const int N = in_sizes[0] / 32;   // 100000
  const int E = in_sizes[1];        // 1600000

  float* ws       = (float*)d_ws;
  float* norm_src = ws;                         // [N]
  float* norm_dst = ws + (size_t)N;             // [N]
  float* h        = ws + (size_t)2 * N;         // [N, 64] capacity
  float* agg      = ws + (size_t)66 * N;        // [N, 64] capacity
  float* z1       = ws + (size_t)130 * N;       // [N, 64]

  const int B = 256;
  const int nMT = N / 16;                       // 6250 exactly
  const int gemmBlocks = (nMT + 7) / 8;         // 8 waves per 256-thread block

  // degrees -> norms (deg_out lives in norm_src, deg_in in norm_dst, transformed in place)
  gcn_zero_kernel<<<512, B, 0, stream>>>(norm_src, (long long)2 * N);
  gcn_degree_kernel<<<(E + B - 1) / B, B, 0, stream>>>(src, dst, norm_src, norm_dst, E);
  gcn_norm_kernel<<<(2 * N + B - 1) / B, B, 0, stream>>>(norm_src, 2 * N);

  // ---- layer 1: 32 -> 64 ----
  gcn_scale_kernel<<<(N * 32 + B - 1) / B, B, 0, stream>>>(x, norm_src, h, N * 32, 5);
  gcn_zero_kernel<<<512, B, 0, stream>>>(agg, (long long)N * 32);
  {
    long long tot = (long long)E * 8;           // d=32 -> 8 float4 chunks per edge
    gcn_scatter_kernel<<<(int)((tot + B - 1) / B), B, 0, stream>>>(h, src, dst, agg, tot, 3, 32);
  }
  gcn_gemm_wmma_kernel<32, 64><<<gemmBlocks, 256, 0, stream>>>(agg, norm_dst, W1, b1, z1, nMT);

  // ---- layer 2: 64 -> 32 ----
  gcn_scale_kernel<<<(N * 64 + B - 1) / B, B, 0, stream>>>(z1, norm_src, h, N * 64, 6);
  gcn_zero_kernel<<<512, B, 0, stream>>>(agg, (long long)N * 64);
  {
    long long tot = (long long)E * 16;          // d=64 -> 16 float4 chunks per edge
    gcn_scatter_kernel<<<(int)((tot + B - 1) / B), B, 0, stream>>>(h, src, dst, agg, tot, 4, 64);
  }
  gcn_gemm_wmma_kernel<64, 32><<<gemmBlocks, 256, 0, stream>>>(agg, norm_dst, W2, b2, out, nMT);
}
